// PointNetPgModel_76931454206587
// MI455X (gfx1250) — matrix operand order repaired and checked by reading, not compiled, not run on hardware
//
#include <hip/hip_runtime.h>
#include <math.h>

typedef __attribute__((ext_vector_type(16))) _Float16 v16h;
typedef __attribute__((ext_vector_type(8)))  _Float16 v8h;
typedef __attribute__((ext_vector_type(8)))  float    v8f;

#define INF_F 3.0e38f

// ---------------------------------------------------------------------------
// Farthest point sampling: one block per cloud, N/256 distances per thread.
// Tie-break: larger distance wins; equal distance -> smaller index (jnp.argmax).
// ---------------------------------------------------------------------------
__global__ void fps_kernel(const float* __restrict__ pos, int N, int m,
                           int* __restrict__ idx) {
  int b = blockIdx.x;
  const float* P = pos + (size_t)b * N * 3;
  int per = N / 256;              // 8 for N=2048, 4 for N=1024
  float d[8];
  float x0 = P[0], y0 = P[1], z0 = P[2];
  for (int j = 0; j < per; ++j) {
    int p = threadIdx.x + j * 256;
    float dx = P[3 * p] - x0, dy = P[3 * p + 1] - y0, dz = P[3 * p + 2] - z0;
    d[j] = dx * dx + dy * dy + dz * dz;
  }
  if (threadIdx.x == 0) idx[(size_t)b * m] = 0;
  __shared__ float sval[256];
  __shared__ int   sidx[256];
  for (int i = 1; i < m; ++i) {
    float bv = -1.0f; int bi = 0x7fffffff;
    for (int j = 0; j < per; ++j) {
      int p = threadIdx.x + j * 256;
      if (d[j] > bv || (d[j] == bv && p < bi)) { bv = d[j]; bi = p; }
    }
    sval[threadIdx.x] = bv; sidx[threadIdx.x] = bi;
    __syncthreads();
    for (int s = 128; s > 0; s >>= 1) {
      if (threadIdx.x < s) {
        if (sval[threadIdx.x + s] > sval[threadIdx.x] ||
            (sval[threadIdx.x + s] == sval[threadIdx.x] &&
             sidx[threadIdx.x + s] < sidx[threadIdx.x])) {
          sval[threadIdx.x] = sval[threadIdx.x + s];
          sidx[threadIdx.x] = sidx[threadIdx.x + s];
        }
      }
      __syncthreads();
    }
    int cur = sidx[0];
    if (threadIdx.x == 0) idx[(size_t)b * m + i] = cur;
    float cx = P[3 * cur], cy = P[3 * cur + 1], cz = P[3 * cur + 2];
    for (int j = 0; j < per; ++j) {
      int p = threadIdx.x + j * 256;
      float dx = P[3 * p] - cx, dy = P[3 * p + 1] - cy, dz = P[3 * p + 2] - cz;
      d[j] = fminf(d[j], dx * dx + dy * dy + dz * dz);
    }
    __syncthreads();
  }
}

__global__ void gather_q(const float* __restrict__ pos, const int* __restrict__ idx,
                         float* __restrict__ q, int N, int m, int total) {
  int i = blockIdx.x * blockDim.x + threadIdx.x;
  if (i >= total) return;
  int b = i / m;
  int p = idx[i];
  const float* src = pos + ((size_t)b * N + p) * 3;
  q[(size_t)i * 3 + 0] = src[0];
  q[(size_t)i * 3 + 1] = src[1];
  q[(size_t)i * 3 + 2] = src[2];
}

__global__ void set_int(int* p, int v) {
  if (blockIdx.x == 0 && threadIdx.x == 0) *p = v;
}

// ---------------------------------------------------------------------------
// Radius ball query with K=64 nearest selection. One wave per query, d^2
// cached in LDS, 64 iterative min extractions (ties -> smaller index).
// ---------------------------------------------------------------------------
__global__ void radius_topk(const float* __restrict__ pos, const float* __restrict__ q,
                            int N, int Mq, int totalQ, float r2,
                            int* __restrict__ nidx, int* __restrict__ mask,
                            int* __restrict__ validCnt) {
  extern __shared__ float sd[];
  int lane = threadIdx.x & 31;
  int wave = threadIdx.x >> 5;
  int wpb  = blockDim.x >> 5;
  int qg = blockIdx.x * wpb + wave;
  if (qg >= totalQ) return;
  float* sdw = sd + (size_t)wave * N;
  int b = qg / Mq;
  const float* P = pos + (size_t)b * N * 3;
  float qx = q[(size_t)qg * 3], qy = q[(size_t)qg * 3 + 1], qz = q[(size_t)qg * 3 + 2];
  int nv = 0;
  for (int p = lane; p < N; p += 32) {
    float dx = P[3 * p] - qx, dy = P[3 * p + 1] - qy, dz = P[3 * p + 2] - qz;
    float dd = dx * dx + dy * dy + dz * dz;
    bool ok = dd <= r2;
    sdw[p] = ok ? dd : INF_F;
    nv += ok ? 1 : 0;
  }
  for (int off = 16; off; off >>= 1) nv += __shfl_xor(nv, off, 32);
  if (lane == 0) atomicAdd(validCnt, nv < 64 ? nv : 64);
  for (int k = 0; k < 64; ++k) {
    __builtin_amdgcn_wave_barrier();
    float bv = INF_F; int bi = N;
    for (int p = lane; p < N; p += 32) {
      float v = sdw[p];
      if (v < bv || (v == bv && p < bi)) { bv = v; bi = p; }
    }
    for (int off = 16; off; off >>= 1) {
      float ov = __shfl_xor(bv, off, 32);
      int   oi = __shfl_xor(bi, off, 32);
      if (ov < bv || (ov == bv && oi < bi)) { bv = ov; bi = oi; }
    }
    if (lane == 0) {
      bool valid = bv < INF_F;
      nidx[(size_t)qg * 64 + k] = valid ? bi : 0;
      mask[(size_t)qg * 64 + k] = valid ? 1 : 0;
      if (valid) sdw[bi] = INF_F;
    }
  }
}

// ---------------------------------------------------------------------------
// Edge-feature builders (f16, zero-padded channel dim for WMMA K granularity)
// ---------------------------------------------------------------------------
__global__ void feat_sa1(const float* __restrict__ pos, const float* __restrict__ q,
                         const int* __restrict__ nidx, _Float16* __restrict__ feat,
                         int N, int Mq, size_t totalElems, int Kp) {
  size_t stride = (size_t)gridDim.x * blockDim.x;
  for (size_t i = (size_t)blockIdx.x * blockDim.x + threadIdx.x; i < totalElems; i += stride) {
    int c = (int)(i % Kp);
    size_t e = i / Kp;
    float v = 0.f;
    if (c < 3) {
      size_t node = e >> 6;
      int b = (int)(node / Mq);
      int j = nidx[e];
      v = pos[((size_t)b * N + j) * 3 + c] - q[node * 3 + c];
    }
    feat[i] = (_Float16)v;
  }
}

__global__ void feat_sa2(const float* __restrict__ x, const float* __restrict__ pos,
                         const float* __restrict__ q, const int* __restrict__ nidx,
                         _Float16* __restrict__ feat, int Cx, int N, int Mq,
                         size_t totalElems, int Kp) {
  size_t stride = (size_t)gridDim.x * blockDim.x;
  for (size_t i = (size_t)blockIdx.x * blockDim.x + threadIdx.x; i < totalElems; i += stride) {
    int c = (int)(i % Kp);
    size_t e = i / Kp;
    size_t node = e >> 6;
    int b = (int)(node / Mq);
    int j = nidx[e];
    float v = 0.f;
    if (c < Cx) v = x[((size_t)b * N + j) * Cx + c];
    else if (c < Cx + 3) v = pos[((size_t)b * N + j) * 3 + (c - Cx)] - q[node * 3 + (c - Cx)];
    feat[i] = (_Float16)v;
  }
}

__global__ void feat_sa3(const float* __restrict__ x, const float* __restrict__ q,
                         _Float16* __restrict__ feat, int Cx, size_t totalElems, int Kp) {
  size_t stride = (size_t)gridDim.x * blockDim.x;
  for (size_t i = (size_t)blockIdx.x * blockDim.x + threadIdx.x; i < totalElems; i += stride) {
    int c = (int)(i % Kp);
    size_t r = i / Kp;
    float v = 0.f;
    if (c < Cx) v = x[r * Cx + c];
    else if (c < Cx + 3) v = q[r * 3 + (c - Cx)];
    feat[i] = (_Float16)v;
  }
}

// Convert W [din,dout] f32 (row-major) -> Wt [dout, Kp] f16 transposed+padded.
__global__ void prep_w(const float* __restrict__ W, _Float16* __restrict__ Wt,
                       int din, int dout, int Kp) {
  size_t total = (size_t)dout * Kp;
  size_t stride = (size_t)gridDim.x * blockDim.x;
  for (size_t i = (size_t)blockIdx.x * blockDim.x + threadIdx.x; i < total; i += stride) {
    int col = (int)(i / Kp);
    int k   = (int)(i % Kp);
    Wt[i] = (_Float16)((k < din) ? W[(size_t)k * dout + col] : 0.f);
  }
}

// ---------------------------------------------------------------------------
// WMMA GEMM, register-blocked along N: each wave computes a 16x64 output
// strip (four 16x16 f32 accumulators), reusing one A fragment per K-chunk
// across 4 v_wmma_f32_16x16x32_f16 ops. Cuts A-side DRAM traffic 4x.
// A fragment: lane l -> row (l&15); halves = K base+(l>=16?16:0)+0..15.
// B fragment: lane l -> col (l&15); halves = K base+(l>=16?16:0)+0..15.
// D: c[r] at lane l -> row tile_m + r + (l>=16?8:0), col tile_n + (l&15).
// Requires Nc % 64 == 0 (true for 64/128/256/512/1024) and Kp % 32 == 0.
// ---------------------------------------------------------------------------
__global__ void gemm_wmma(const _Float16* __restrict__ A, const _Float16* __restrict__ Wt,
                          const float* __restrict__ bias, _Float16* __restrict__ H,
                          int Mrows, int Nc, int Kp) {
  int lane = threadIdx.x & 31;
  int wave = threadIdx.x >> 5;
  int tile_n = blockIdx.x * 64;
  int tile_m = (blockIdx.y * (blockDim.x >> 5) + wave) * 16;
  if (tile_m >= Mrows) return;
  int row = tile_m + (lane & 15);
  int klo_a = (lane >> 4) * 16;
  int klo_b = (lane >> 4) * 16;
  const _Float16* Arow = A + (size_t)row * Kp;
  const _Float16* Bcol = Wt + (size_t)(tile_n + (lane & 15)) * Kp;
  v8f acc0 = {}, acc1 = {}, acc2 = {}, acc3 = {};
  for (int kc = 0; kc < Kp; kc += 32) {
    v16h a = *(const v16h*)(Arow + kc + klo_a);
    v16h bf0 = *(const v16h*)(Bcol + kc + klo_b);
    v16h bf1 = *(const v16h*)(Bcol + (size_t)16 * Kp + kc + klo_b);
    v16h bf2 = *(const v16h*)(Bcol + (size_t)32 * Kp + kc + klo_b);
    v16h bf3 = *(const v16h*)(Bcol + (size_t)48 * Kp + kc + klo_b);
    acc0 = __builtin_amdgcn_wmma_f32_16x16x32_f16(false, a, false, bf0,
                                                  (short)0, acc0, false, false);
    acc1 = __builtin_amdgcn_wmma_f32_16x16x32_f16(false, a, false, bf1,
                                                  (short)0, acc1, false, false);
    acc2 = __builtin_amdgcn_wmma_f32_16x16x32_f16(false, a, false, bf2,
                                                  (short)0, acc2, false, false);
    acc3 = __builtin_amdgcn_wmma_f32_16x16x32_f16(false, a, false, bf3,
                                                  (short)0, acc3, false, false);
  }
  int rbase = tile_m + (lane >> 4) * 8;
  int c0 = tile_n + (lane & 15);
#pragma unroll
  for (int t = 0; t < 4; ++t) {
    int col = c0 + t * 16;
    float bb = bias[col];
    v8f acc = (t == 0) ? acc0 : (t == 1) ? acc1 : (t == 2) ? acc2 : acc3;
#pragma unroll
    for (int r = 0; r < 8; ++r) {
      H[((size_t)(rbase + r)) * Nc + col] = (_Float16)(acc[r] + bb);
    }
  }
}

// ---------------------------------------------------------------------------
// Masked batch-norm statistics: one block per channel.
// ---------------------------------------------------------------------------
__global__ void bn_stats(const _Float16* __restrict__ H, const int* __restrict__ mask,
                         int E, int C, float* __restrict__ sum, float* __restrict__ sumsq) {
  int c = blockIdx.x;
  float s = 0.f, s2 = 0.f;
  for (int r = threadIdx.x; r < E; r += blockDim.x) {
    if (!mask || mask[r]) {
      float v = (float)H[(size_t)r * C + c];
      s += v; s2 += v * v;
    }
  }
  __shared__ float as_[256];
  __shared__ float bs_[256];
  as_[threadIdx.x] = s; bs_[threadIdx.x] = s2;
  __syncthreads();
  for (int st = 128; st > 0; st >>= 1) {
    if (threadIdx.x < st) {
      as_[threadIdx.x] += as_[threadIdx.x + st];
      bs_[threadIdx.x] += bs_[threadIdx.x + st];
    }
    __syncthreads();
  }
  if (threadIdx.x == 0) { sum[c] = as_[0]; sumsq[c] = bs_[0]; }
}

__global__ void bn_relu(_Float16* __restrict__ H, const float* __restrict__ sum,
                        const float* __restrict__ sumsq, const int* __restrict__ cntPtr,
                        const float* __restrict__ g, const float* __restrict__ beta,
                        size_t total, int C) {
  float cnt = (float)(*cntPtr);
  size_t stride = (size_t)gridDim.x * blockDim.x;
  for (size_t i = (size_t)blockIdx.x * blockDim.x + threadIdx.x; i < total; i += stride) {
    int c = (int)(i % C);
    float mean = sum[c] / cnt;
    float var = sumsq[c] / cnt - mean * mean;
    float v = ((float)H[i] - mean) * rsqrtf(var + 1e-5f) * g[c] + beta[c];
    H[i] = (_Float16)fmaxf(v, 0.f);
  }
}

// Masked max over the K=64 neighbor axis.
__global__ void max_agg(const _Float16* __restrict__ H, const int* __restrict__ mask,
                        float* __restrict__ out, size_t nodes, int C) {
  size_t total = nodes * C;
  size_t stride = (size_t)gridDim.x * blockDim.x;
  for (size_t i = (size_t)blockIdx.x * blockDim.x + threadIdx.x; i < total; i += stride) {
    size_t node = i / C;
    int c = (int)(i % C);
    float best = -INF_F;
    for (int k = 0; k < 64; ++k) {
      if (mask[node * 64 + k]) {
        float v = (float)H[((node << 6) + k) * (size_t)C + c];
        best = fmaxf(best, v);
      }
    }
    out[i] = best;
  }
}

__global__ void global_max(const _Float16* __restrict__ H, float* __restrict__ g,
                           int Bn, int P, int C) {
  size_t total = (size_t)Bn * C;
  size_t stride = (size_t)gridDim.x * blockDim.x;
  for (size_t i = (size_t)blockIdx.x * blockDim.x + threadIdx.x; i < total; i += stride) {
    int b = (int)(i / C);
    int c = (int)(i % C);
    float best = -INF_F;
    for (int p = 0; p < P; ++p)
      best = fmaxf(best, (float)H[((size_t)b * P + p) * C + c]);
    g[i] = best;
  }
}

// Small f32 linear (heads): the reference heads have NO ReLU and NO norm.
__global__ void linear_small(const float* __restrict__ X, const float* __restrict__ W,
                             const float* __restrict__ bias, float* __restrict__ Y,
                             int M, int Kd, int Nc) {
  int i = blockIdx.x * blockDim.x + threadIdx.x;
  if (i >= M * Nc) return;
  int m = i / Nc, n = i % Nc;
  float s = bias[n];
  for (int k = 0; k < Kd; ++k) s += X[(size_t)m * Kd + k] * W[(size_t)k * Nc + n];
  Y[i] = s;
}

__global__ void softmax16(const float* __restrict__ logits, float* __restrict__ out,
                          int Mrows) {
  int r = blockIdx.x * blockDim.x + threadIdx.x;
  if (r >= Mrows) return;
  float mx = -INF_F;
  for (int c = 0; c < 16; ++c) mx = fmaxf(mx, logits[r * 16 + c]);
  float e[16], s = 0.f;
  for (int c = 0; c < 16; ++c) { e[c] = expf(logits[r * 16 + c] - mx); s += e[c]; }
  for (int c = 0; c < 16; ++c) out[r * 16 + c] = e[c] / s;
}

// ---------------------------------------------------------------------------
extern "C" void kernel_launch(void* const* d_in, const int* in_sizes, int n_in,
                              void* d_out, int out_size, void* d_ws, size_t ws_size,
                              hipStream_t stream) {
  (void)in_sizes; (void)n_in; (void)out_size; (void)ws_size;
  const int B = 8, N1 = 2048, M1 = 1024, N2 = 1024, M2 = 256;
  const int E1 = B * M1 * 64;      // 524288
  const int E2 = B * M2 * 64;      // 131072
  const int ROWS3 = B * M2;        // 2048
  const int Kp1 = 32, Kp2 = 160, Kp3 = 288;

  const float* pos = (const float*)d_in[0];
  const float *W11 = (const float*)d_in[1],  *b11 = (const float*)d_in[2];
  const float *g11 = (const float*)d_in[3],  *be11 = (const float*)d_in[4];
  const float *W12 = (const float*)d_in[5],  *b12 = (const float*)d_in[6];
  const float *g12 = (const float*)d_in[7],  *be12 = (const float*)d_in[8];
  const float *W13 = (const float*)d_in[9],  *b13 = (const float*)d_in[10];
  const float *W21 = (const float*)d_in[11], *b21 = (const float*)d_in[12];
  const float *g21 = (const float*)d_in[13], *be21 = (const float*)d_in[14];
  const float *W22 = (const float*)d_in[15], *b22 = (const float*)d_in[16];
  const float *g22 = (const float*)d_in[17], *be22 = (const float*)d_in[18];
  const float *W23 = (const float*)d_in[19], *b23 = (const float*)d_in[20];
  const float *W31 = (const float*)d_in[21], *b31 = (const float*)d_in[22];
  const float *g31 = (const float*)d_in[23], *be31 = (const float*)d_in[24];
  const float *W32 = (const float*)d_in[25], *b32 = (const float*)d_in[26];
  const float *g32 = (const float*)d_in[27], *be32 = (const float*)d_in[28];
  const float *W33 = (const float*)d_in[29], *b33 = (const float*)d_in[30];
  const float *Wp0 = (const float*)d_in[31], *bp0 = (const float*)d_in[32];
  const float *Wp1 = (const float*)d_in[33], *bp1 = (const float*)d_in[34];
  const float *Wp2 = (const float*)d_in[35], *bp2 = (const float*)d_in[36];
  const float *Wv0 = (const float*)d_in[37], *bv0 = (const float*)d_in[38];
  const float *Wv1 = (const float*)d_in[39], *bv1 = (const float*)d_in[40];
  const float *Wv2 = (const float*)d_in[41], *bv2 = (const float*)d_in[42];

  // ---- workspace carve ----
  char* base = (char*)d_ws;
  size_t off = 0;
  auto alloc = [&](size_t bytes) -> char* {
    char* p = base + off;
    off += (bytes + 255) & ~(size_t)255;
    return p;
  };
  _Float16* bufA = (_Float16*)alloc((size_t)E1 * 64 * 2);    // 64 MiB ping
  _Float16* bufB = (_Float16*)alloc((size_t)E1 * 128 * 2);   // 128 MiB pong
  int*   idx1  = (int*)alloc((size_t)B * M1 * 4);
  float* q1    = (float*)alloc((size_t)B * M1 * 3 * 4);
  int*   nidx1 = (int*)alloc((size_t)E1 * 4);
  int*   mask1 = (int*)alloc((size_t)E1 * 4);
  int*   cnt1  = (int*)alloc(4);
  float* x1    = (float*)alloc((size_t)B * M1 * 128 * 4);
  int*   idx2  = (int*)alloc((size_t)B * M2 * 4);
  float* q2    = (float*)alloc((size_t)B * M2 * 3 * 4);
  int*   nidx2 = (int*)alloc((size_t)E2 * 4);
  int*   mask2 = (int*)alloc((size_t)E2 * 4);
  int*   cnt2  = (int*)alloc(4);
  int*   cnt3  = (int*)alloc(4);
  float* x2    = (float*)alloc((size_t)B * M2 * 256 * 4);
  float* gbuf  = (float*)alloc((size_t)B * 1024 * 4);
  float* ssum  = (float*)alloc(1024 * 4);
  float* ssq   = (float*)alloc(1024 * 4);
  _Float16* Wt11 = (_Float16*)alloc((size_t)64 * Kp1 * 2);
  _Float16* Wt12 = (_Float16*)alloc((size_t)64 * 64 * 2);
  _Float16* Wt13 = (_Float16*)alloc((size_t)128 * 64 * 2);
  _Float16* Wt21 = (_Float16*)alloc((size_t)128 * Kp2 * 2);
  _Float16* Wt22 = (_Float16*)alloc((size_t)128 * 128 * 2);
  _Float16* Wt23 = (_Float16*)alloc((size_t)256 * 128 * 2);
  _Float16* Wt31 = (_Float16*)alloc((size_t)256 * Kp3 * 2);
  _Float16* Wt32 = (_Float16*)alloc((size_t)512 * 256 * 2);
  _Float16* Wt33 = (_Float16*)alloc((size_t)1024 * 512 * 2);
  float* t1 = (float*)alloc((size_t)8 * 512 * 4);
  float* t2 = (float*)alloc((size_t)8 * 256 * 4);
  float* lg = (float*)alloc((size_t)8 * 16 * 4);

  auto blks = [](size_t n) { return (unsigned)((n + 255) / 256); };

  // ---- weight prep (transpose + pad + f16) ----
  prep_w<<<blks(64 * Kp1), 256, 0, stream>>>(W11, Wt11, 3, 64, Kp1);
  prep_w<<<blks(64 * 64), 256, 0, stream>>>(W12, Wt12, 64, 64, 64);
  prep_w<<<blks(128 * 64), 256, 0, stream>>>(W13, Wt13, 64, 128, 64);
  prep_w<<<blks(128 * Kp2), 256, 0, stream>>>(W21, Wt21, 131, 128, Kp2);
  prep_w<<<blks(128 * 128), 256, 0, stream>>>(W22, Wt22, 128, 128, 128);
  prep_w<<<blks(256 * 128), 256, 0, stream>>>(W23, Wt23, 128, 256, 128);
  prep_w<<<blks(256 * Kp3), 256, 0, stream>>>(W31, Wt31, 259, 256, Kp3);
  prep_w<<<blks(512 * 256), 256, 0, stream>>>(W32, Wt32, 256, 512, 256);
  prep_w<<<blks(1024 * 512), 256, 0, stream>>>(W33, Wt33, 512, 1024, 512);

  // ---- SA1 ----
  fps_kernel<<<B, 256, 0, stream>>>(pos, N1, M1, idx1);
  gather_q<<<blks(B * M1), 256, 0, stream>>>(pos, idx1, q1, N1, M1, B * M1);
  set_int<<<1, 1, 0, stream>>>(cnt1, 0);
  radius_topk<<<(B * M1) / 4, 128, 4 * N1 * 4, stream>>>(pos, q1, N1, M1, B * M1,
                                                         0.04f, nidx1, mask1, cnt1);
  feat_sa1<<<8192, 256, 0, stream>>>(pos, q1, nidx1, bufA, N1, M1,
                                     (size_t)E1 * Kp1, Kp1);
  gemm_wmma<<<dim3(1, E1 / 128), 256, 0, stream>>>(bufA, Wt11, b11, bufB, E1, 64, Kp1);
  bn_stats<<<64, 256, 0, stream>>>(bufB, mask1, E1, 64, ssum, ssq);
  bn_relu<<<8192, 256, 0, stream>>>(bufB, ssum, ssq, cnt1, g11, be11,
                                    (size_t)E1 * 64, 64);
  gemm_wmma<<<dim3(1, E1 / 128), 256, 0, stream>>>(bufB, Wt12, b12, bufA, E1, 64, 64);
  bn_stats<<<64, 256, 0, stream>>>(bufA, mask1, E1, 64, ssum, ssq);
  bn_relu<<<8192, 256, 0, stream>>>(bufA, ssum, ssq, cnt1, g12, be12,
                                    (size_t)E1 * 64, 64);
  gemm_wmma<<<dim3(2, E1 / 128), 256, 0, stream>>>(bufA, Wt13, b13, bufB, E1, 128, 64);
  max_agg<<<4096, 256, 0, stream>>>(bufB, mask1, x1, (size_t)B * M1, 128);

  // ---- SA2 ----
  fps_kernel<<<B, 256, 0, stream>>>(q1, N2, M2, idx2);
  gather_q<<<blks(B * M2), 256, 0, stream>>>(q1, idx2, q2, N2, M2, B * M2);
  set_int<<<1, 1, 0, stream>>>(cnt2, 0);
  radius_topk<<<(B * M2) / 4, 128, 4 * N2 * 4, stream>>>(q1, q2, N2, M2, B * M2,
                                                         0.16f, nidx2, mask2, cnt2);
  feat_sa2<<<8192, 256, 0, stream>>>(x1, q1, q2, nidx2, bufA, 128, N2, M2,
                                     (size_t)E2 * Kp2, Kp2);
  gemm_wmma<<<dim3(2, E2 / 128), 256, 0, stream>>>(bufA, Wt21, b21, bufB, E2, 128, Kp2);
  bn_stats<<<128, 256, 0, stream>>>(bufB, mask2, E2, 128, ssum, ssq);
  bn_relu<<<4096, 256, 0, stream>>>(bufB, ssum, ssq, cnt2, g21, be21,
                                    (size_t)E2 * 128, 128);
  gemm_wmma<<<dim3(2, E2 / 128), 256, 0, stream>>>(bufB, Wt22, b22, bufA, E2, 128, 128);
  bn_stats<<<128, 256, 0, stream>>>(bufA, mask2, E2, 128, ssum, ssq);
  bn_relu<<<4096, 256, 0, stream>>>(bufA, ssum, ssq, cnt2, g22, be22,
                                    (size_t)E2 * 128, 128);
  gemm_wmma<<<dim3(4, E2 / 128), 256, 0, stream>>>(bufA, Wt23, b23, bufB, E2, 256, 128);
  max_agg<<<2048, 256, 0, stream>>>(bufB, mask2, x2, (size_t)B * M2, 256);

  // ---- SA3 (global MLP, unmasked BN over all B*256 rows) ----
  set_int<<<1, 1, 0, stream>>>(cnt3, ROWS3);
  feat_sa3<<<blks((size_t)ROWS3 * Kp3), 256, 0, stream>>>(x2, q2, bufA, 256,
                                                          (size_t)ROWS3 * Kp3, Kp3);
  gemm_wmma<<<dim3(4, ROWS3 / 128), 256, 0, stream>>>(bufA, Wt31, b31, bufB,
                                                      ROWS3, 256, Kp3);
  bn_stats<<<256, 256, 0, stream>>>(bufB, nullptr, ROWS3, 256, ssum, ssq);
  bn_relu<<<2048, 256, 0, stream>>>(bufB, ssum, ssq, cnt3, g31, be31,
                                    (size_t)ROWS3 * 256, 256);
  gemm_wmma<<<dim3(8, ROWS3 / 128), 256, 0, stream>>>(bufB, Wt32, b32, bufA,
                                                      ROWS3, 512, 256);
  bn_stats<<<512, 256, 0, stream>>>(bufA, nullptr, ROWS3, 512, ssum, ssq);
  bn_relu<<<4096, 256, 0, stream>>>(bufA, ssum, ssq, cnt3, g32, be32,
                                    (size_t)ROWS3 * 512, 512);
  gemm_wmma<<<dim3(16, ROWS3 / 128), 256, 0, stream>>>(bufA, Wt33, b33, bufB,
                                                       ROWS3, 1024, 512);
  global_max<<<32, 256, 0, stream>>>(bufB, gbuf, B, M2, 1024);

  // ---- heads (pure linear stacks, then softmax for pi) ----
  float* out = (float*)d_out;
  linear_small<<<blks(8 * 512), 256, 0, stream>>>(gbuf, Wp0, bp0, t1, 8, 1024, 512);
  linear_small<<<blks(8 * 256), 256, 0, stream>>>(t1, Wp1, bp1, t2, 8, 512, 256);
  linear_small<<<blks(8 * 16), 256, 0, stream>>>(t2, Wp2, bp2, lg, 8, 256, 16);
  softmax16<<<1, 32, 0, stream>>>(lg, out, 8);
  linear_small<<<blks(8 * 512), 256, 0, stream>>>(gbuf, Wv0, bv0, t1, 8, 1024, 512);
  linear_small<<<blks(8 * 256), 256, 0, stream>>>(t1, Wv1, bv1, t2, 8, 512, 256);
  linear_small<<<blks(8 * 1), 256, 0, stream>>>(t2, Wv2, bv2, out + 128, 8, 256, 1);
}
